// MultiGNN_11269994185511
// MI455X (gfx1250) — matrix operand order repaired and checked by reading, not compiled, not run on hardware
//
#include <hip/hip_runtime.h>
#include <hip/hip_bf16.h>

typedef __attribute__((ext_vector_type(16))) __bf16 v16bf;
typedef __attribute__((ext_vector_type(8)))  float  v8f;
typedef __attribute__((ext_vector_type(4)))  unsigned int v4u;

namespace {

constexpr int LDSW = 40;   // row stride in ushorts (80B): 16B-aligned rows, staggers banks

// f32 -> bf16 (RNE), pure 32-bit ALU (avoids v_mov_b16 subregister lowering)
__device__ __forceinline__ unsigned bfround(float f) {
    union { float f; unsigned u; } v; v.f = f;
    return v.u + 0x7FFFu + ((v.u >> 16) & 1u);      // bf16 in top 16 bits
}
__device__ __forceinline__ unsigned pk2(float lo, float hi) {
    return (bfround(lo) >> 16) | (bfround(hi) & 0xFFFF0000u);
}

union Frag { v16bf v; v4u q[2]; };

// ---------------- bf16 x bf16 -> f32 WMMA GEMM ----------------
// C[M,N] = act( A[M,K] @ B[K,N] + bias ), A/B in bf16, C f32.
// Block tile 128x64, 8 wave32s, each wave 32x32 = 2x2 v_wmma_f32_16x16x32_bf16.
// Double-buffered LDS. REQUIRES K % 8 == 0 and N % 8 == 0 (true: K in {512,800,400},
// N in {800,200,64}) so 8-element bf16 chunks are never partially valid.
template<bool RELU, bool BIAS, bool GATHER>
__global__ __launch_bounds__(256)
void wmma_gemm(const unsigned short* __restrict__ A, const unsigned short* __restrict__ B,
               const float* __restrict__ bias, const int* __restrict__ pairs,
               float* __restrict__ C, int M, int N, int K, int Dhalf)
{
    __shared__ __align__(16) unsigned short As[2][128 * LDSW];
    __shared__ __align__(16) unsigned short Bs[2][64 * LDSW];

    const int tid    = threadIdx.x;
    const int lane   = tid & 31;
    const int laneLo = lane & 15;
    const int laneHi = lane >> 4;          // 0 / 1
    const int wave   = tid >> 5;           // 0..7
    const int waveM  = wave >> 1;          // 0..3
    const int waveN  = wave & 1;           // 0..1
    const int blockM0 = blockIdx.x * 128;
    const int blockN0 = blockIdx.y * 64;

    // per-thread staging coordinates (loop-invariant)
    const int ar_r[2]  = { (tid + 0)   >> 2, (tid + 256) >> 2 };      // A rows 0..127
    const int ar_c8    = (tid & 3) << 3;                               // A K-offset 0..24
    const int b_kk  = tid >> 3;                                        // 0..31
    const int b_nn8 = (tid & 7) << 3;                                  // 0..56

    const v8f zacc = {0.f,0.f,0.f,0.f,0.f,0.f,0.f,0.f};
    v8f acc[2][2];
    #pragma unroll
    for (int i = 0; i < 2; ++i)
        #pragma unroll
        for (int j = 0; j < 2; ++j)
            acc[i][j] = zacc;

    const int nkt = (K + 31) >> 5;

    uint4 aq[2], bq;                       // staged registers (raw bf16 payload)
    bool aok[2], bok;                      // OOB flags, applied at *store* time so the
                                           // loadcnt waits sink below the WMMAs

    auto load_tiles = [&](int k0) {
        #pragma unroll
        for (int i = 0; i < 2; ++i) {      // A tile 128x32: 8 bf16 (16B) per chunk
            const int grow = blockM0 + ar_r[i];
            const int gk   = k0 + ar_c8;
            const bool ok  = (grow < M) && (gk < K);
            const int growc = (grow < M) ? grow : (M - 1);
            const int gkc   = ok ? gk : 0;
            if (GATHER) {                  // A row = h[pairs[t, gk>=Dhalf]] (Dhalf%8==0)
                const int half = (gkc >= Dhalf) ? 1 : 0;
                const int node = pairs[2 * growc + half];
                aq[i] = *(const uint4*)&A[(size_t)node * Dhalf + (gkc - half * Dhalf)];
            } else {
                aq[i] = *(const uint4*)&A[(size_t)growc * K + gkc];
            }
            aok[i] = ok;
        }
        {                                  // B tile 32x64: 8 bf16 (16B) per thread
            const int gk = k0 + b_kk;
            const int gn = blockN0 + b_nn8;
            bok = (gk < K) && (gn < N);
            const int gkc = (gk < K) ? gk : (K - 1);
            const int gnc = bok ? gn : 0;
            bq = *(const uint4*)&B[(size_t)gkc * N + gnc];
        }
    };

    auto store_tiles = [&](unsigned short* __restrict__ Ad, unsigned short* __restrict__ Bd) {
        #pragma unroll
        for (int i = 0; i < 2; ++i) {      // A: zero-fix then ds_store_b128
            uint4 q = aq[i];
            if (!aok[i]) { q.x = 0u; q.y = 0u; q.z = 0u; q.w = 0u; }
            *(uint4*)&Ad[ar_r[i] * LDSW + ar_c8] = q;
        }
        uint4 q = bq;
        if (!bok) { q.x = 0u; q.y = 0u; q.z = 0u; q.w = 0u; }
        // B: transposed b16 scatter Bd[n][k]
        Bd[(b_nn8 + 0) * LDSW + b_kk] = (unsigned short)(q.x & 0xFFFFu);
        Bd[(b_nn8 + 1) * LDSW + b_kk] = (unsigned short)(q.x >> 16);
        Bd[(b_nn8 + 2) * LDSW + b_kk] = (unsigned short)(q.y & 0xFFFFu);
        Bd[(b_nn8 + 3) * LDSW + b_kk] = (unsigned short)(q.y >> 16);
        Bd[(b_nn8 + 4) * LDSW + b_kk] = (unsigned short)(q.z & 0xFFFFu);
        Bd[(b_nn8 + 5) * LDSW + b_kk] = (unsigned short)(q.z >> 16);
        Bd[(b_nn8 + 6) * LDSW + b_kk] = (unsigned short)(q.w & 0xFFFFu);
        Bd[(b_nn8 + 7) * LDSW + b_kk] = (unsigned short)(q.w >> 16);
    };

    load_tiles(0);
    store_tiles(As[0], Bs[0]);

    for (int kt = 0; kt < nkt; ++kt) {
        __syncthreads();                   // staged buffer [kt&1] now visible
        const int cur = kt & 1;
        const bool more = (kt + 1) < nkt;

        if (more) load_tiles((kt + 1) << 5);   // global b128 loads overlap ds+wmma

        // fragment loads per ISA VGPR layouts (2x ds_load_b128 each)
        Frag aF[2], bF[2];
        #pragma unroll
        for (int i = 0; i < 2; ++i) {
            const int row = waveM * 32 + i * 16 + laneLo;
            const int ko  = laneHi << 3;                            // hi lanes start K=8
            aF[i].q[0] = *(const v4u*)&As[cur][row * LDSW + ko];      // K {0..7}/{8..15}
            aF[i].q[1] = *(const v4u*)&As[cur][row * LDSW + ko + 16]; // K {16..23}/{24..31}
        }
        #pragma unroll
        for (int j = 0; j < 2; ++j) {
            const int col = waveN * 32 + j * 16 + laneLo;
            const int ko  = laneHi << 3;                            // hi lanes start K=8
            bF[j].q[0] = *(const v4u*)&Bs[cur][col * LDSW + ko];
            bF[j].q[1] = *(const v4u*)&Bs[cur][col * LDSW + ko + 16];
        }

        #pragma unroll
        for (int i = 0; i < 2; ++i)
            #pragma unroll
            for (int j = 0; j < 2; ++j)
                acc[i][j] = __builtin_amdgcn_wmma_f32_16x16x32_bf16(
                    false, aF[i].v, false, bF[j].v, (short)0, acc[i][j], false, false);

        if (more) store_tiles(As[cur ^ 1], Bs[cur ^ 1]);   // write other buffer, no barrier
    }

    // epilogue: v8f elem r -> row (laneHi?8:0)+r, col laneLo.
    // One base address per fragment; advance by N per row (no per-element 64-bit mul).
    #pragma unroll
    for (int i = 0; i < 2; ++i) {
        const int rBase    = blockM0 + waveM * 32 + i * 16 + laneHi * 8;
        const int rowsLeft = M - rBase;                 // rows valid in this fragment
        #pragma unroll
        for (int j = 0; j < 2; ++j) {
            const int col = blockN0 + waveN * 32 + j * 16 + laneLo;
            if (col < N) {
                const float bv = BIAS ? bias[col] : 0.f;
                float* __restrict__ cp = C + (size_t)rBase * N + col;
                #pragma unroll
                for (int r = 0; r < 8; ++r) {
                    if (r < rowsLeft) {
                        float v = acc[i][j][r] + bv;
                        if (RELU) v = fmaxf(v, 0.f);
                        *cp = v;
                    }
                    cp += N;
                }
            }
        }
    }
}

// ---------------- elementwise helpers (all sizes % 4 == 0) ----------------

__global__ void to_bf16(const float* __restrict__ s, unsigned short* __restrict__ d, int n4) {
    const int i = blockIdx.x * 256 + threadIdx.x;
    if (i >= n4) return;
    const float4 q = *(const float4*)&s[(size_t)i * 4];
    uint2 p; p.x = pk2(q.x, q.y); p.y = pk2(q.z, q.w);
    *(uint2*)&d[(size_t)i * 4] = p;
}

__global__ void relu_to_bf16(const float* __restrict__ s, unsigned short* __restrict__ d, int n4) {
    const int i = blockIdx.x * 256 + threadIdx.x;
    if (i >= n4) return;
    const float4 q = *(const float4*)&s[(size_t)i * 4];
    uint2 p;
    p.x = pk2(fmaxf(q.x, 0.f), fmaxf(q.y, 0.f));
    p.y = pk2(fmaxf(q.z, 0.f), fmaxf(q.w, 0.f));
    *(uint2*)&d[(size_t)i * 4] = p;
}

__global__ void combine_to_bf16(const float* __restrict__ a, const float* __restrict__ b,
                                unsigned short* __restrict__ d, int n4) {
    const int i = blockIdx.x * 256 + threadIdx.x;
    if (i >= n4) return;
    const float4 qa = *(const float4*)&a[(size_t)i * 4];
    const float4 qb = *(const float4*)&b[(size_t)i * 4];
    uint2 p;                                   // (1-λ)h1 + λh2, λ=0.5
    p.x = pk2(0.5f * (qa.x + qb.x), 0.5f * (qa.y + qb.y));
    p.y = pk2(0.5f * (qa.z + qb.z), 0.5f * (qa.w + qb.w));
    *(uint2*)&d[(size_t)i * 4] = p;
}

__global__ void zero_f32(float* p, int n4) {
    const int i = blockIdx.x * 256 + threadIdx.x;
    if (i >= n4) return;
    float4 z; z.x = 0.f; z.y = 0.f; z.z = 0.f; z.w = 0.f;
    *(float4*)&p[(size_t)i * 4] = z;
}

__global__ void relu_f32(float* p, int n4) {
    const int i = blockIdx.x * 256 + threadIdx.x;
    if (i >= n4) return;
    float4 q = *(const float4*)&p[(size_t)i * 4];
    q.x = fmaxf(q.x, 0.f); q.y = fmaxf(q.y, 0.f);
    q.z = fmaxf(q.z, 0.f); q.w = fmaxf(q.w, 0.f);
    *(float4*)&p[(size_t)i * 4] = q;
}

// Y[rows[e], :] += vals[e] * X[cols[e], :]
__global__ void spmm_atomic(const float* __restrict__ vals, const int* __restrict__ rows,
                            const int* __restrict__ cols, const float* __restrict__ X,
                            float* __restrict__ Y, int E, int Dim) {
    const int e = blockIdx.x;
    const int d = blockIdx.y * blockDim.x + threadIdx.x;
    if (e >= E || d >= Dim) return;
    const float v = vals[e];
    const int r = rows[e], c = cols[e];
    atomicAdd(&Y[(size_t)r * Dim + d], v * X[(size_t)c * Dim + d]);
}

// out[t] = relu( Xh[t,:] . w2 ),  K = 64
__global__ void mlp_out(const float* __restrict__ Xh, const float* __restrict__ w2,
                        float* __restrict__ out, int T) {
    __shared__ float w[64];
    if (threadIdx.x < 64) w[threadIdx.x] = w2[threadIdx.x];
    __syncthreads();
    const int t = blockIdx.x * blockDim.x + threadIdx.x;
    if (t >= T) return;
    const float4* x = (const float4*)&Xh[(size_t)t * 64];
    float s = 0.f;
    #pragma unroll
    for (int j = 0; j < 16; ++j) {
        const float4 q = x[j];
        s = fmaf(q.x, w[4*j+0], s); s = fmaf(q.y, w[4*j+1], s);
        s = fmaf(q.z, w[4*j+2], s); s = fmaf(q.w, w[4*j+3], s);
    }
    out[t] = fmaxf(s, 0.f);
}

} // anonymous namespace

extern "C" void kernel_launch(void* const* d_in, const int* in_sizes, int n_in,
                              void* d_out, int out_size, void* d_ws, size_t ws_size,
                              hipStream_t stream) {
    (void)in_sizes; (void)n_in; (void)out_size; (void)ws_size;
    constexpr int Nn = 20000, F = 512, H1 = 800, D = 200, E = 160000, T = 131072, MH = 64;

    const float* rna    = (const float*)d_in[0];
    const float* atac   = (const float*)d_in[1];
    const int*   rows   = (const int*)  d_in[2];
    const int*   cols   = (const int*)  d_in[3];
    const float* vals   = (const float*)d_in[4];
    const int*   pairs  = (const int*)  d_in[5];   // train_sample [T,2]
    const float* w_rna1 = (const float*)d_in[6];
    const float* w_rna2 = (const float*)d_in[7];
    const float* w_atac1= (const float*)d_in[8];
    const float* w_atac2= (const float*)d_in[9];
    const float* mlp_w1 = (const float*)d_in[10];
    const float* mlp_b1 = (const float*)d_in[11];
    const float* mlp_w2 = (const float*)d_in[12];
    float* out = (float*)d_out;

    // ---- workspace carve-up: f32 region then bf16 region (~293 MB total) ----
    float* tmpA = (float*)d_ws;                    // N*H1  gemm1 out (spmm1 input)
    float* tmpB = tmpA + (size_t)Nn * H1;          // N*H1  spmm1 accum
    float* tmpC = tmpB + (size_t)Nn * H1;          // N*D   gemm2 out (spmm2 input)
    float* h1   = tmpC + (size_t)Nn * D;           // N*D
    float* h2   = h1   + (size_t)Nn * D;           // N*D
    float* Xh   = h2   + (size_t)Nn * D;           // T*MH  MLP hidden
    unsigned short* rnaH  = (unsigned short*)(Xh + (size_t)T * MH);  // N*F
    unsigned short* atacH = rnaH  + (size_t)Nn * F;                  // N*F
    unsigned short* w1H0  = atacH + (size_t)Nn * F;                  // F*H1
    unsigned short* w1H1  = w1H0  + (size_t)F * H1;                  // F*H1
    unsigned short* w2H0  = w1H1  + (size_t)F * H1;                  // H1*D
    unsigned short* w2H1  = w2H0  + (size_t)H1 * D;                  // H1*D
    unsigned short* mw1H  = w2H1  + (size_t)H1 * D;                  // 2D*MH
    unsigned short* tmpBh = mw1H  + (size_t)2 * D * MH;              // N*H1
    unsigned short* hcH   = tmpBh + (size_t)Nn * H1;                 // N*D

    const dim3 blk(256);
    const dim3 gemm1_grid((Nn + 127) / 128, (H1 + 63) / 64);
    const dim3 gemm2_grid((Nn + 127) / 128, (D  + 63) / 64);
    const dim3 gemm3_grid((T  + 127) / 128, (MH + 63) / 64);
    const dim3 spmm1_grid(E, (H1 + 255) / 256);
    const dim3 spmm2_grid(E, (D  + 255) / 256);

    // ---- one-time (per launch) f32 -> bf16 conversion of GEMM operands ----
    auto cvt = [&](const float* s, unsigned short* d, size_t n) {
        const int n4 = (int)(n / 4);
        to_bf16<<<(n4 + 255) / 256, blk, 0, stream>>>(s, d, n4);
    };
    cvt(rna,     rnaH,  (size_t)Nn * F);
    cvt(atac,    atacH, (size_t)Nn * F);
    cvt(w_rna1,  w1H0,  (size_t)F * H1);
    cvt(w_atac1, w1H1,  (size_t)F * H1);
    cvt(w_rna2,  w2H0,  (size_t)H1 * D);
    cvt(w_atac2, w2H1,  (size_t)H1 * D);
    cvt(mlp_w1,  mw1H,  (size_t)2 * D * MH);

    for (int tower = 0; tower < 2; ++tower) {
        const unsigned short* featH = tower ? atacH : rnaH;
        const unsigned short* w1H   = tower ? w1H1  : w1H0;
        const unsigned short* w2H   = tower ? w2H1  : w2H0;
        float* hT                   = tower ? h2    : h1;

        // tmpA = feat @ w1 ; tmpB = spmm(tmpA) ; tmpBh = bf16(relu(tmpB))
        wmma_gemm<false,false,false><<<gemm1_grid, blk, 0, stream>>>(
            featH, w1H, nullptr, nullptr, tmpA, Nn, H1, F, 0);
        zero_f32<<<((Nn * H1 / 4) + 255) / 256, blk, 0, stream>>>(tmpB, Nn * H1 / 4);
        spmm_atomic<<<spmm1_grid, blk, 0, stream>>>(vals, rows, cols, tmpA, tmpB, E, H1);
        relu_to_bf16<<<((Nn * H1 / 4) + 255) / 256, blk, 0, stream>>>(tmpB, tmpBh, Nn * H1 / 4);

        // tmpC = h @ w2 ; hT = relu(spmm(tmpC))
        wmma_gemm<false,false,false><<<gemm2_grid, blk, 0, stream>>>(
            tmpBh, w2H, nullptr, nullptr, tmpC, Nn, D, H1, 0);
        zero_f32<<<((Nn * D / 4) + 255) / 256, blk, 0, stream>>>(hT, Nn * D / 4);
        spmm_atomic<<<spmm2_grid, blk, 0, stream>>>(vals, rows, cols, tmpC, hT, E, D);
        relu_f32<<<((Nn * D / 4) + 255) / 256, blk, 0, stream>>>(hT, Nn * D / 4);
    }

    // hcH = bf16( 0.5*h1 + 0.5*h2 )
    combine_to_bf16<<<((Nn * D / 4) + 255) / 256, blk, 0, stream>>>(h1, h2, hcH, Nn * D / 4);

    // Xh = relu( concat(h[src], h[dst]) @ mlp_w1 + b1 )  -- gather fused into A-load
    wmma_gemm<true,true,true><<<gemm3_grid, blk, 0, stream>>>(
        hcH, mw1H, mlp_b1, pairs, Xh, T, MH, 2 * D, D);

    // out = relu( Xh @ mlp_w2 )
    mlp_out<<<(T + 255) / 256, blk, 0, stream>>>(Xh, mlp_w2, out, T);
}
